// BiLSTM_CRF_39745627357371
// MI455X (gfx1250) — matrix operand order repaired and checked by reading, not compiled, not run on hardware
//
#include <hip/hip_runtime.h>
#include <hip/hip_bf16.h>
#include <math.h>

#define E_       512
#define NLAYERS_ 2
#define NL_      9
#define B_       64
#define T_       512
#define H_       256
#define BT_      (B_ * T_)   // 32768

typedef __bf16 bf16_t;
typedef __bf16 v16bf __attribute__((ext_vector_type(16)));
typedef __bf16 v8bf  __attribute__((ext_vector_type(8)));
typedef float  v8f   __attribute__((ext_vector_type(8)));

union V16 { v16bf v; v8bf h[2]; };

__device__ __forceinline__ float sigf(float x) { return 1.0f / (1.0f + __expf(-x)); }

// ---------------- TDM availability (device pass only; fallback compiles everywhere) ----
#ifdef __HIP_DEVICE_COMPILE__
#  if __has_builtin(__builtin_amdgcn_tensor_load_to_lds) && __has_builtin(__builtin_amdgcn_s_wait_tensorcnt)
#    define HAVE_TDM 1
#  else
#    define HAVE_TDM 0
#  endif
#else
#  define HAVE_TDM 0
#endif

#if HAVE_TDM
typedef unsigned int u32x4 __attribute__((ext_vector_type(4)));
typedef int          i32x8 __attribute__((ext_vector_type(8)));
typedef int          i32x4 __attribute__((ext_vector_type(4)));

// 2D TDM load: tile (K=32 contiguous) x (M=128 rows), 2-byte elements,
// tensor row length/stride = 512 elements, packed row-major into LDS.
__device__ __forceinline__ void tdm_load_tile_32x128(unsigned int lds_addr, const bf16_t* gptr) {
    unsigned long long ga = (unsigned long long)(uintptr_t)gptr;
    u32x4 g0;
    g0[0] = 1u;                                            // count=1, user descriptor
    g0[1] = lds_addr;                                      // LDS byte address
    g0[2] = (unsigned int)ga;                              // global_addr[31:0]
    g0[3] = (unsigned int)((ga >> 32) & 0x01FFFFFFu) | (2u << 30);  // addr[56:32] | type=2
    i32x8 g1;
    g1[0] = (int)(1u << 16);                               // wg_mask=0 | data_size=1 (2B)
    g1[1] = (int)(((unsigned int)512 & 0xFFFFu) << 16);    // tensor_dim0[15:0]=512
    g1[2] = (int)(((unsigned int)BT_ & 0xFFFFu) << 16);    // dim0 hi=0 | tensor_dim1[15:0]
    g1[3] = (int)(32u << 16);                              // dim1 hi=0 | tile_dim0=32
    g1[4] = 128;                                           // tile_dim1=128 | tile_dim2=0
    g1[5] = 512;                                           // tensor_dim0_stride[31:0]=512
    g1[6] = 0;                                             // stride0 hi | stride1 lo
    g1[7] = 0;                                             // stride1 hi
    i32x4 z4 = {0, 0, 0, 0};
#  if __clang_major__ >= 23
    i32x8 z8 = {0, 0, 0, 0, 0, 0, 0, 0};
    __builtin_amdgcn_tensor_load_to_lds(g0, g1, z4, z4, z8, 0);
#  else
    __builtin_amdgcn_tensor_load_to_lds(g0, g1, z4, z4, 0);
#  endif
}
#endif // HAVE_TDM

// ---------------- elementwise fp32 -> bf16 cast ----------------
__global__ void k_cast_bf16(const float* __restrict__ src, bf16_t* __restrict__ dst, int n) {
    int i = blockIdx.x * blockDim.x + threadIdx.x;
    if (i < n) dst[i] = (bf16_t)src[i];
}

// ---------------- embedding gather -> bf16 ----------------
__global__ void k_embed(const int* __restrict__ ids, const float* __restrict__ emb,
                        bf16_t* __restrict__ x) {
    int i  = blockIdx.x * blockDim.x + threadIdx.x;   // over BT_*512
    int bt = i >> 9;
    int e  = i & 511;
    int row = ids[bt];
    x[i] = (bf16_t)emb[(size_t)row * E_ + e];
}

// ---------------- WMMA GEMM: XP = X(BT x 512) * W^T(512 x 1024) + (b_ih + b_hh) ----
// Block: 256 threads = 8 waves, arranged 2(M-groups) x 4(N-groups).
// Wave tile: 64M x 64N (4x4 WMMA tiles). Block tile: 128M x 256N.
// A tile (128x32 bf16, 8 KB) is DMA'd to LDS by the Tensor Data Mover, double-buffered.
// grid = (BT/128, 1024/256).
__global__ __launch_bounds__(256) void k_gemm_xw(
        const bf16_t* __restrict__ X,    // (BT, 512) row-major bf16
        const bf16_t* __restrict__ W,    // (1024, 512) row-major bf16 (gate-major)
        const float*  __restrict__ bih,  // (1024,)
        const float*  __restrict__ bhh,  // (1024,)
        float* __restrict__ XP)          // (BT, 1024)
{
    const int lane   = threadIdx.x & 31;
    const int wave   = threadIdx.x >> 5;       // 0..7
    const int nlo    = lane & 15;
    const int hi     = lane >> 4;
    const int mgrp   = wave >> 2;              // 0..1
    const int ngrp   = wave & 3;               // 0..3
    const int mblock = blockIdx.x * 128;
    const int nbase  = blockIdx.y * 256 + ngrp * 64;

    v8f acc[4][4];
#pragma unroll
    for (int mt = 0; mt < 4; ++mt)
#pragma unroll
        for (int nt = 0; nt < 4; ++nt)
#pragma unroll
            for (int r = 0; r < 8; ++r) acc[mt][nt][r] = 0.0f;

#if HAVE_TDM
    __shared__ bf16_t abuf[2][128 * 32];       // double-buffered A tile (2 x 8 KB)
    const unsigned int lds0 = (unsigned int)(uintptr_t)&abuf[0][0];
    const unsigned int lds1 = (unsigned int)(uintptr_t)&abuf[1][0];
    if (threadIdx.x == 0)
        tdm_load_tile_32x128(lds0, X + (size_t)mblock * 512);
#endif

    for (int ks = 0; ks < 16; ++ks) {
        const int k0 = ks * 32;
#if HAVE_TDM
        if (threadIdx.x == 0)
            __builtin_amdgcn_s_wait_tensorcnt(0);     // buffer ks resident in LDS
        __syncthreads();                              // publish to all waves
        if (threadIdx.x == 0 && ks < 15)              // prefetch next tile into other buffer
            tdm_load_tile_32x128((ks & 1) ? lds0 : lds1,
                                 X + (size_t)mblock * 512 + (k0 + 32));
        const bf16_t* abase = abuf[ks & 1];
#endif
        // B fragments (32x16): lane n holds 16 contiguous K values of W row (nbase+nt*16+n)
        v16bf bfr[4];
#pragma unroll
        for (int nt = 0; nt < 4; ++nt)
            bfr[nt] = *(const v16bf*)(W + (size_t)(nbase + nt * 16 + nlo) * 512 + k0 + 16 * hi);

#pragma unroll
        for (int mt = 0; mt < 4; ++mt) {
            V16 a;
#if HAVE_TDM
            const bf16_t* hp = abase + (mgrp * 64 + mt * 16 + nlo) * 32 + 16 * hi;
            a.h[0] = *(const v8bf*)(hp);          // K = k0 + 16*hi + [0..7]
            a.h[1] = *(const v8bf*)(hp + 8);      // K = k0 + 16*hi + 8 + [0..7]
#else
            const bf16_t* xrow =
                X + (size_t)(mblock + mgrp * 64 + mt * 16 + nlo) * 512 + k0 + 16 * hi;
            a.h[0] = *(const v8bf*)(xrow);
            a.h[1] = *(const v8bf*)(xrow + 8);
#endif
#pragma unroll
            for (int nt = 0; nt < 4; ++nt)
                acc[mt][nt] = __builtin_amdgcn_wmma_f32_16x16x32_bf16(
                    false, a.v, false, bfr[nt], (short)0, acc[mt][nt], false, false);
        }
    }

#pragma unroll
    for (int nt = 0; nt < 4; ++nt) {
        int col = nbase + nt * 16 + nlo;
        float bias = bih[col] + bhh[col];
#pragma unroll
        for (int mt = 0; mt < 4; ++mt) {
#pragma unroll
            for (int r = 0; r < 8; ++r) {
                int row = mblock + mgrp * 64 + mt * 16 + r + 8 * hi;
                XP[(size_t)row * 1024 + col] = acc[mt][nt][r] + bias;
            }
        }
    }
}

// ---------------- recurrent LSTM scan (one layer, both dirs in grid.y) -------------
// Block: 512 threads = 16 waves; wave w owns hidden cols [16w, 16w+16) => gate tiles
// {w, 16+w, 32+w, 48+w} (i,f,g,o). W_hh held resident in VGPRs, h in LDS, c in regs.
__global__ __launch_bounds__(512) void k_lstm_rec(
        const bf16_t* __restrict__ WHH,   // (2, 1024, 256) bf16, this layer
        const float*  __restrict__ XPb,   // (2, BT, 1024) fp32, biases folded in
        bf16_t* __restrict__ Xnext)       // (BT, 512); dir writes cols [dir*256, +256)
{
    const int dir   = blockIdx.y;
    const int bbase = blockIdx.x * 16;        // batch rows [bbase, bbase+16)
    const int lane  = threadIdx.x & 31;
    const int w     = threadIdx.x >> 5;       // 0..15
    const int nlo   = lane & 15;
    const int hi    = lane >> 4;

    const bf16_t* wh = WHH + (size_t)dir * 1024 * 256;
    const float*  xp = XPb + (size_t)dir * BT_ * 1024;

    __shared__ bf16_t hsh[16 * 256];          // h_{t-1}: 16 batch rows x 256 units (8 KB)

    // Preload all B fragments of W_hh for this wave's 4 gate tiles (resident in VGPRs)
    v16bf bfrag[4][8];
#pragma unroll
    for (int g = 0; g < 4; ++g) {
        int gcol = g * 256 + w * 16 + nlo;    // gate row of W_hh
#pragma unroll
        for (int ks = 0; ks < 8; ++ks)
            bfrag[g][ks] = *(const v16bf*)(wh + (size_t)gcol * 256 + ks * 32 + 16 * hi);
    }

    for (int i = threadIdx.x; i < 16 * 256; i += 512) hsh[i] = (bf16_t)0.0f;
    v8f c;
#pragma unroll
    for (int r = 0; r < 8; ++r) c[r] = 0.0f;
    __syncthreads();

    for (int s = 0; s < T_; ++s) {
        const int t = dir ? (T_ - 1 - s) : s;

        // Read A fragments (h_{t-1}, 16x256) from LDS
        V16 a[8];
#pragma unroll
        for (int ks = 0; ks < 8; ++ks) {
            const bf16_t* hp = hsh + nlo * 256 + ks * 32 + 16 * hi;
            a[ks].h[0] = *(const v8bf*)(hp);
            a[ks].h[1] = *(const v8bf*)(hp + 8);
        }

        // Init accumulators from precomputed input projection xp[b, t, gate]
        v8f acc[4];
#pragma unroll
        for (int g = 0; g < 4; ++g) {
            int col = g * 256 + w * 16 + nlo;
#pragma unroll
            for (int r = 0; r < 8; ++r) {
                int brow = bbase + r + 8 * hi;
                acc[g][r] = xp[(size_t)(brow * T_ + t) * 1024 + col];
            }
        }

        // Prefetch next step's gate pre-activations (hide L2/HBM latency behind WMMAs)
        if (s + 1 < T_) {
            const int t2 = dir ? (T_ - 2 - s) : (s + 1);
#pragma unroll
            for (int g = 0; g < 4; ++g) {
                int col = g * 256 + w * 16 + nlo;
                __builtin_prefetch(&xp[(size_t)((bbase + 8 * hi) * T_ + t2) * 1024 + col], 0, 0);
                __builtin_prefetch(&xp[(size_t)((bbase + 4 + 8 * hi) * T_ + t2) * 1024 + col], 0, 0);
            }
        }
        __syncthreads();   // all LDS reads of h_{t-1} done before rewriting

#pragma unroll
        for (int g = 0; g < 4; ++g)
#pragma unroll
            for (int ks = 0; ks < 8; ++ks)
                acc[g] = __builtin_amdgcn_wmma_f32_16x16x32_bf16(
                    false, a[ks].v, false, bfrag[g][ks], (short)0, acc[g], false, false);

        // Activations, cell/h update; write h to LDS + layer output
#pragma unroll
        for (int r = 0; r < 8; ++r) {
            float ig = sigf(acc[0][r]);
            float fg = sigf(acc[1][r]);
            float gg = tanhf(acc[2][r]);
            float og = sigf(acc[3][r]);
            float cn = fg * c[r] + ig * gg;
            c[r] = cn;
            float hv = og * tanhf(cn);
            int m   = r + 8 * hi;
            int col = w * 16 + nlo;
            bf16_t hb = (bf16_t)hv;
            hsh[m * 256 + col] = hb;
            Xnext[(size_t)((bbase + m) * T_ + t) * 512 + dir * 256 + col] = hb;
        }
        __syncthreads();   // h_t fully written before next step's reads
    }
}

// ---------------- classifier: logits = x @ cls_w^T + cls_b  (N=9, VALU) ------------
__global__ void k_cls(const bf16_t* __restrict__ X, const float* __restrict__ cw,
                      const float* __restrict__ cb, float* __restrict__ logits) {
    int i = blockIdx.x * blockDim.x + threadIdx.x;
    if (i >= BT_ * NL_) return;
    int bt = i / NL_, l = i % NL_;
    const bf16_t* xr = X + (size_t)bt * 512;
    const float*  wr = cw + (size_t)l * 512;
    float acc = 0.0f;
    for (int k = 0; k < 512; ++k) acc += (float)xr[k] * wr[k];
    logits[i] = acc + cb[l];
}

// ---------------- CRF negative log-likelihood (64 independent chains) --------------
__global__ __launch_bounds__(64) void k_crf(const float* __restrict__ logits,
        const int* __restrict__ labels, const float* __restrict__ st,
        const float* __restrict__ en, const float* __restrict__ tr,
        float* __restrict__ out) {
    __shared__ float s_tr[NL_ * NL_];
    __shared__ float s_st[NL_], s_en[NL_];
    __shared__ float partial[64];
    int tid = threadIdx.x;
    if (tid < NL_ * NL_) s_tr[tid] = tr[tid];
    if (tid < NL_) { s_st[tid] = st[tid]; s_en[tid] = en[tid]; }
    __syncthreads();

    const int b = tid;
    const float* lg = logits + (size_t)b * T_ * NL_;
    const int*   lb = labels + (size_t)b * T_;
    float alpha[NL_];
    int prev = lb[0];
    float num = s_st[prev] + lg[prev];
    for (int j = 0; j < NL_; ++j) alpha[j] = s_st[j] + lg[j];

    for (int t = 1; t < T_; ++t) {
        const float* em = lg + t * NL_;
        float na[NL_];
        for (int j = 0; j < NL_; ++j) {
            float mx = -1e30f;
            for (int i2 = 0; i2 < NL_; ++i2)
                mx = fmaxf(mx, alpha[i2] + s_tr[i2 * NL_ + j]);
            float ssum = 0.0f;
            for (int i2 = 0; i2 < NL_; ++i2)
                ssum += __expf(alpha[i2] + s_tr[i2 * NL_ + j] - mx);
            na[j] = mx + __logf(ssum) + em[j];
        }
        for (int j = 0; j < NL_; ++j) alpha[j] = na[j];
        int y = lb[t];
        num += s_tr[prev * NL_ + y] + em[y];
        prev = y;
    }
    float mx = -1e30f;
    for (int j = 0; j < NL_; ++j) mx = fmaxf(mx, alpha[j] + s_en[j]);
    float ssum = 0.0f;
    for (int j = 0; j < NL_; ++j) ssum += __expf(alpha[j] + s_en[j] - mx);
    float denom = mx + __logf(ssum);
    num += s_en[prev];

    partial[tid] = denom - num;
    __syncthreads();
    if (tid == 0) {
        float s = 0.0f;
        for (int i2 = 0; i2 < 64; ++i2) s += partial[i2];
        out[0] = s;   // -sum(num - denom)
    }
}

// ---------------- host orchestration ----------------
extern "C" void kernel_launch(void* const* d_in, const int* in_sizes, int n_in,
                              void* d_out, int out_size, void* d_ws, size_t ws_size,
                              hipStream_t stream) {
    (void)in_sizes; (void)n_in; (void)out_size; (void)ws_size;
    const int*   input_ids = (const int*)d_in[0];
    const int*   labels    = (const int*)d_in[1];
    const float* emb       = (const float*)d_in[2];
    const float* w_ih      = (const float*)d_in[3];  // (2,2,1024,512)
    const float* w_hh      = (const float*)d_in[4];  // (2,2,1024,256)
    const float* b_ih      = (const float*)d_in[5];  // (2,2,1024)
    const float* b_hh      = (const float*)d_in[6];
    const float* cls_w     = (const float*)d_in[7];
    const float* cls_b     = (const float*)d_in[8];
    const float* st        = (const float*)d_in[9];
    const float* en        = (const float*)d_in[10];
    const float* tr        = (const float*)d_in[11];
    float* out = (float*)d_out;

    char* ws = (char*)d_ws;
    size_t off = 0;
    auto alloc = [&](size_t bytes) -> char* {
        char* p = ws + off;
        off += (bytes + 255) & ~(size_t)255;
        return p;
    };
    bf16_t* xA   = (bf16_t*)alloc((size_t)BT_ * 512 * 2);            // 32 MiB
    bf16_t* xB   = (bf16_t*)alloc((size_t)BT_ * 512 * 2);            // 32 MiB
    float*  xp   = (float*) alloc((size_t)2 * BT_ * 1024 * 4);       // 256 MiB (both dirs)
    bf16_t* wihb = (bf16_t*)alloc((size_t)2 * 2 * 1024 * 512 * 2);   // 4 MiB
    bf16_t* whhb = (bf16_t*)alloc((size_t)2 * 2 * 1024 * 256 * 2);   // 2 MiB
    float*  lgts = (float*) alloc((size_t)BT_ * NL_ * 4);            // ~1.1 MiB

    {
        int n = 2 * 2 * 1024 * 512;
        k_cast_bf16<<<(n + 255) / 256, 256, 0, stream>>>(w_ih, wihb, n);
        n = 2 * 2 * 1024 * 256;
        k_cast_bf16<<<(n + 255) / 256, 256, 0, stream>>>(w_hh, whhb, n);
    }
    k_embed<<<(BT_ * 512) / 256, 256, 0, stream>>>(input_ids, emb, xA);

    bf16_t* xin = xA;
    bf16_t* xout = xB;
    for (int l = 0; l < NLAYERS_; ++l) {
        for (int d = 0; d < 2; ++d) {
            const bf16_t* W  = wihb + (size_t)(l * 2 + d) * 1024 * 512;
            const float*  bi = b_ih + (size_t)(l * 2 + d) * 1024;
            const float*  bh = b_hh + (size_t)(l * 2 + d) * 1024;
            float* xpd = xp + (size_t)d * BT_ * 1024;
            dim3 g(BT_ / 128, 4);
            k_gemm_xw<<<g, 256, 0, stream>>>(xin, W, bi, bh, xpd);
        }
        dim3 gr(4, 2);
        k_lstm_rec<<<gr, 512, 0, stream>>>(whhb + (size_t)l * 2 * 1024 * 256, xp, xout);
        bf16_t* tmp = xin; xin = xout; xout = tmp;
    }
    k_cls<<<(BT_ * NL_ + 255) / 256, 256, 0, stream>>>(xin, cls_w, cls_b, lgts);
    k_crf<<<1, 64, 0, stream>>>(lgts, labels, st, en, tr, out);
}
